// RetBlock_11828339933935
// MI455X (gfx1250) — matrix operand order repaired and checked by reading, not compiled, not run on hardware
//
#include <hip/hip_runtime.h>
#include <hip/hip_bf16.h>

typedef __bf16  bf16x16 __attribute__((ext_vector_type(16)));
typedef float   floatx8 __attribute__((ext_vector_type(8)));
typedef unsigned short u16;

union Frag { bf16x16 v; uint4 q[2]; };

__device__ __forceinline__ u16 f2bf(float f) {
  unsigned int u = __float_as_uint(f);
  u += 0x7fffu + ((u >> 16) & 1u);          // round-to-nearest-even
  return (u16)(u >> 16);
}

// ---------------------------------------------------------------------------
// Generic depthwise conv (SAME padding): out = [addsrc +] conv(in) + bias
// in/addsrc addressed as [pixel*inStride + inOfs + c]; outputs dense [pixel*Cc+c]
// ---------------------------------------------------------------------------
__global__ void dwconv_kernel(const float* __restrict__ in, const float* __restrict__ w,
                              const float* __restrict__ bias, const float* __restrict__ addsrc,
                              float* __restrict__ outF, u16* __restrict__ outBf,
                              int n, int H, int W, int Cc, int R, int inStride, int inOfs)
{
  int i = blockIdx.x * blockDim.x + threadIdx.x;
  if (i >= n) return;
  int c   = i % Cc;
  int pix = i / Cc;
  int wx  = pix % W;
  int hy  = (pix / W) % H;
  int pb  = pix / (W * H);
  int D   = 2 * R + 1;
  float acc = bias[c];
  for (int kh = 0; kh < D; ++kh) {
    int yy = hy + kh - R;
    if (yy < 0 || yy >= H) continue;
    for (int kw = 0; kw < D; ++kw) {
      int xx = wx + kw - R;
      if (xx < 0 || xx >= W) continue;
      int p2 = (pb * H + yy) * W + xx;
      acc += in[(size_t)p2 * inStride + inOfs + c] * w[(kh * D + kw) * Cc + c];
    }
  }
  if (addsrc) acc += addsrc[(size_t)pix * inStride + inOfs + c];
  if (outF)  outF [(size_t)pix * Cc + c] = acc;
  if (outBf) outBf[(size_t)pix * Cc + c] = f2bf(acc);
}

// ---------------------------------------------------------------------------
// LayerNorm over C=384, output bf16
// ---------------------------------------------------------------------------
__global__ __launch_bounds__(128) void ln_kernel(const float* __restrict__ x,
                                                 const float* __restrict__ g,
                                                 const float* __restrict__ b,
                                                 u16* __restrict__ out)
{
  const int C = 384;
  int row = blockIdx.x, t = threadIdx.x;
  __shared__ float rs[128], rq[128];
  const float* xr = x + (size_t)row * C;
  float v0 = xr[t], v1 = xr[t + 128], v2 = xr[t + 256];
  rs[t] = v0 + v1 + v2;
  rq[t] = v0*v0 + v1*v1 + v2*v2;
  __syncthreads();
  for (int o = 64; o > 0; o >>= 1) {
    if (t < o) { rs[t] += rs[t + o]; rq[t] += rq[t + o]; }
    __syncthreads();
  }
  float mean = rs[0] * (1.0f / 384.0f);
  float var  = rq[0] * (1.0f / 384.0f) - mean * mean;
  float rstd = rsqrtf(var + 1e-6f);
  u16* orow = out + (size_t)row * C;
  orow[t]       = f2bf((v0 - mean) * rstd * g[t]       + b[t]);
  orow[t + 128] = f2bf((v1 - mean) * rstd * g[t + 128] + b[t + 128]);
  orow[t + 256] = f2bf((v2 - mean) * rstd * g[t + 256] + b[t + 256]);
}

// ---------------------------------------------------------------------------
// fp32 weight [K,N] -> bf16 TRANSPOSED [Ntot,K] (with scale, row offset).
// Pre-transposing at conversion time makes GEMM B-tile staging pure b128
// LDS stores (no scalar transpose scatter in the hot loop).
// ---------------------------------------------------------------------------
__global__ void convert_wT_kernel(const float* __restrict__ src, u16* __restrict__ dst,
                                  int K, int N, int rowOfs, float scale)
{
  int i = blockIdx.x * blockDim.x + threadIdx.x;
  if (i >= K * N) return;
  int k = i / N, nn = i % N;
  dst[(size_t)(rowOfs + nn) * K + k] = f2bf(src[i] * scale);
}

__global__ void pack_bias_kernel(const float* __restrict__ src, float* __restrict__ dst,
                                 int n, int ofs, float scale)
{
  int i = blockIdx.x * blockDim.x + threadIdx.x;
  if (i < n) dst[ofs + i] = src[i] * scale;
}

// ---------------------------------------------------------------------------
// Tiled bf16 WMMA GEMM: C[M,N] = act(A[M,K] @ B[K,N] + bias) [+ residual]
// B supplied pre-transposed as Bt[N,K]. Block tile 64x128, 8 waves (2x4),
// each wave 32x32 via 2x2 wmma frags, K step 32. M%64==0, N%128==0, K%32==0.
// ---------------------------------------------------------------------------
__global__ __launch_bounds__(256) void gemm_kernel(
    const u16* __restrict__ A, const u16* __restrict__ Bt,
    const float* __restrict__ bias, const float* __restrict__ residual,
    float* __restrict__ Cout, int M, int N, int K, int act)
{
  __shared__ __align__(16) u16 As[64][32];
  __shared__ __align__(16) u16 Bs[128][32];   // [n][k]
  int tid  = threadIdx.x;
  int lane = tid & 31, wave = tid >> 5;
  int lr = lane & 15,  hi   = lane >> 4;
  int bm = blockIdx.x * 64, bn = blockIdx.y * 128;
  int wm = (wave >> 2) * 32, wn = (wave & 3) * 32;
  floatx8 acc[2][2] = {};

  int arow = tid >> 2, acol = (tid & 3) * 8;    // A: 64x32, 8 halfs/thread
  int brow = tid >> 1, bcol = (tid & 1) * 16;   // Bt: 128x32, 16 halfs/thread

  const u16* aptr = A  + (size_t)(bm + arow) * K + acol;
  const u16* bptr = Bt + (size_t)(bn + brow) * K + bcol;

  for (int k0 = 0; k0 < K; k0 += 32) {
    *(uint4*)&As[arow][acol] = *(const uint4*)(aptr + k0);
    {
      uint4 p0 = *(const uint4*)(bptr + k0);
      uint4 p1 = *(const uint4*)(bptr + k0 + 8);
      *(uint4*)&Bs[brow][bcol]     = p0;
      *(uint4*)&Bs[brow][bcol + 8] = p1;
    }
    if (k0 + 32 < K) {
      __builtin_prefetch(aptr + k0 + 32, 0, 1);
      __builtin_prefetch(bptr + k0 + 32, 0, 1);
    }
    __syncthreads();
    Frag af[2], bfr[2];
#pragma unroll
    for (int i = 0; i < 2; ++i) {           // A frag: lanes 0-15 rows, K split hi*8 / 16+hi*8
      const u16* p = &As[wm + i * 16 + lr][hi * 8];
      af[i].q[0] = *(const uint4*)p;
      af[i].q[1] = *(const uint4*)(p + 16);
    }
#pragma unroll
    for (int j = 0; j < 2; ++j) {           // B frag: col = lr, K split hi*8 / 16+hi*8
      const u16* p = &Bs[wn + j * 16 + lr][hi * 8];
      bfr[j].q[0] = *(const uint4*)p;
      bfr[j].q[1] = *(const uint4*)(p + 16);
    }
#pragma unroll
    for (int i = 0; i < 2; ++i)
#pragma unroll
      for (int j = 0; j < 2; ++j)
        acc[i][j] = __builtin_amdgcn_wmma_f32_16x16x32_bf16(
            false, af[i].v, false, bfr[j].v, (short)0, acc[i][j], false, false);
    __syncthreads();
  }

#pragma unroll
  for (int i = 0; i < 2; ++i)
#pragma unroll
    for (int j = 0; j < 2; ++j)
#pragma unroll
      for (int r = 0; r < 8; ++r) {
        int row = bm + wm + i * 16 + r + hi * 8;
        int col = bn + wn + j * 16 + lr;
        float v = acc[i][j][r] + bias[col];
        if (act) v = 0.5f * v * (1.0f + erff(v * 0.70710678f));   // exact GELU
        if (residual) v += residual[(size_t)row * N + col];
        Cout[(size_t)row * N + col] = v;
      }
}

// ---------------------------------------------------------------------------
// Rotary (theta_shift) + layout change: qkv fp32 [M,1152] ->
//   qr,kr bf16 [B,NH,L,HD]  and  vT bf16 [B,NH,HD,L]
// ---------------------------------------------------------------------------
__global__ void rotary_kernel(const float* __restrict__ qkv,
                              const float* __restrict__ sinb, const float* __restrict__ cosb,
                              u16* __restrict__ qr, u16* __restrict__ kr, u16* __restrict__ vT)
{
  const int NH = 12, L = 1024, HD = 32;
  int i = blockIdx.x * blockDim.x + threadIdx.x;
  if (i >= 4 * L * NH * (HD / 2)) return;
  int p   = i % 16;   int rem = i / 16;
  int nh  = rem % NH; rem /= NH;
  int l   = rem % L;  int b = rem / L;
  int row = b * L + l;
  int c0  = nh * HD + 2 * p;
  float s0 = sinb[l * HD + 2 * p],     s1 = sinb[l * HD + 2 * p + 1];
  float cc0 = cosb[l * HD + 2 * p],    cc1 = cosb[l * HD + 2 * p + 1];
  const float* qp = qkv + (size_t)row * 1152;
  float q0 = qp[c0],       q1 = qp[c0 + 1];
  float k0 = qp[384 + c0], k1 = qp[384 + c0 + 1];
  float v0 = qp[768 + c0], v1 = qp[768 + c0 + 1];
  size_t hbase = ((size_t)(b * NH + nh)) * L * HD;
  qr[hbase + (size_t)l * HD + 2 * p]     = f2bf(q0 * cc0 - q1 * s0);
  qr[hbase + (size_t)l * HD + 2 * p + 1] = f2bf(q1 * cc1 + q0 * s1);
  kr[hbase + (size_t)l * HD + 2 * p]     = f2bf(k0 * cc0 - k1 * s0);
  kr[hbase + (size_t)l * HD + 2 * p + 1] = f2bf(k1 * cc1 + k0 * s1);
  size_t tbase = ((size_t)(b * NH + nh)) * HD * L;
  vT[tbase + (size_t)(2 * p)     * L + l] = f2bf(v0);
  vT[tbase + (size_t)(2 * p + 1) * L + l] = f2bf(v1);
}

// ---------------------------------------------------------------------------
// Fused flash-style attention: one wave per (b, nh, 16-row q tile).
// Mask recomputed as decay[nh]*manhattan(l,m) (decay read from mask[nh,0,1]),
// saving ~192 MB of HBM traffic. Output: bf16 (o/l + lepe) into obf [M,384].
// ---------------------------------------------------------------------------
__global__ __launch_bounds__(32) void attn_kernel(
    const u16* __restrict__ qr, const u16* __restrict__ kr, const u16* __restrict__ vT,
    const float* __restrict__ mask, const float* __restrict__ lepe, u16* __restrict__ obf)
{
  const int L = 1024, HD = 32, NH = 12, W = 32;
  int b = blockIdx.z, nh = blockIdx.y, q0 = blockIdx.x * 16;
  int lane = threadIdx.x;
  int lr = lane & 15, hi = lane >> 4;
  const u16* qh = qr + ((size_t)(b * NH + nh)) * L * HD;
  const u16* kh = kr + ((size_t)(b * NH + nh)) * L * HD;
  const u16* vh = vT + ((size_t)(b * NH + nh)) * HD * L;
  float decay = mask[(size_t)nh * L * L + 1];     // dist(0,1)==1
  __shared__ __align__(16) u16 Ps[16][32];

  Frag aq;
  {
    const u16* p = qh + (size_t)(q0 + lr) * HD + hi * 8;
    aq.q[0] = *(const uint4*)p;
    aq.q[1] = *(const uint4*)(p + 16);
  }
  floatx8 oacc[2] = {};
  float mrun[8], lrun[8];
#pragma unroll
  for (int r = 0; r < 8; ++r) { mrun[r] = -1e30f; lrun[r] = 0.0f; }

  int myrow[8], ry[8], rx[8];
#pragma unroll
  for (int r = 0; r < 8; ++r) {
    myrow[r] = q0 + r + hi * 8;
    ry[r] = myrow[r] >> 5; rx[r] = myrow[r] & (W - 1);
  }

  for (int kc = 0; kc < L; kc += 32) {
    Frag bk[2];
#pragma unroll
    for (int j = 0; j < 2; ++j) {
      const u16* p = kh + (size_t)(kc + j * 16 + lr) * HD + hi * 8;
      bk[j].q[0] = *(const uint4*)p;
      bk[j].q[1] = *(const uint4*)(p + 16);
    }
    floatx8 z{};
    floatx8 s[2];
    s[0] = __builtin_amdgcn_wmma_f32_16x16x32_bf16(false, aq.v, false, bk[0].v, (short)0, z, false, false);
    s[1] = __builtin_amdgcn_wmma_f32_16x16x32_bf16(false, aq.v, false, bk[1].v, (short)0, z, false, false);

    int key0 = kc + lr, key1 = kc + 16 + lr;
    int k0y = key0 >> 5, k0x = key0 & (W - 1);
    int k1y = key1 >> 5, k1x = key1 & (W - 1);
    float mx[8];
#pragma unroll
    for (int r = 0; r < 8; ++r) {
      float m0 = decay * (float)(abs(ry[r] - k0y) + abs(rx[r] - k0x));
      float m1 = decay * (float)(abs(ry[r] - k1y) + abs(rx[r] - k1x));
      s[0][r] += m0; s[1][r] += m1;
      float v = fmaxf(s[0][r], s[1][r]);
      v = fmaxf(v, __shfl_xor(v, 1, 32));
      v = fmaxf(v, __shfl_xor(v, 2, 32));
      v = fmaxf(v, __shfl_xor(v, 4, 32));
      v = fmaxf(v, __shfl_xor(v, 8, 32));
      mx[r] = v;
    }
#pragma unroll
    for (int r = 0; r < 8; ++r) {
      float mnew = fmaxf(mrun[r], mx[r]);
      float corr = __expf(mrun[r] - mnew);
      mrun[r] = mnew;
      float p0 = __expf(s[0][r] - mnew);
      float p1 = __expf(s[1][r] - mnew);
      float t = p0 + p1;
      t += __shfl_xor(t, 1, 32);
      t += __shfl_xor(t, 2, 32);
      t += __shfl_xor(t, 4, 32);
      t += __shfl_xor(t, 8, 32);
      lrun[r] = lrun[r] * corr + t;
      oacc[0][r] *= corr; oacc[1][r] *= corr;
      int row = r + hi * 8;                 // C-layout -> LDS
      Ps[row][lr]      = f2bf(p0);
      Ps[row][16 + lr] = f2bf(p1);
    }
    __syncthreads();
    Frag pa;                                // reload as A-layout fragment
    {
      const u16* p = &Ps[lr][hi * 8];
      pa.q[0] = *(const uint4*)p;
      pa.q[1] = *(const uint4*)(p + 16);
    }
    Frag bv[2];
#pragma unroll
    for (int j2 = 0; j2 < 2; ++j2) {
      const u16* p = vh + (size_t)(j2 * 16 + lr) * L + kc + hi * 8;
      bv[j2].q[0] = *(const uint4*)p;
      bv[j2].q[1] = *(const uint4*)(p + 16);
    }
    oacc[0] = __builtin_amdgcn_wmma_f32_16x16x32_bf16(false, pa.v, false, bv[0].v, (short)0, oacc[0], false, false);
    oacc[1] = __builtin_amdgcn_wmma_f32_16x16x32_bf16(false, pa.v, false, bv[1].v, (short)0, oacc[1], false, false);
    __syncthreads();
  }

#pragma unroll
  for (int j2 = 0; j2 < 2; ++j2)
#pragma unroll
    for (int r = 0; r < 8; ++r) {
      int row = myrow[r];
      int cch = nh * HD + j2 * 16 + lr;
      size_t idx = ((size_t)(b * L) + row) * 384 + cch;
      float val = oacc[j2][r] / lrun[r] + lepe[idx];
      obf[idx] = f2bf(val);
    }
}

// ---------------------------------------------------------------------------
extern "C" void kernel_launch(void* const* d_in, const int* in_sizes, int n_in,
                              void* d_out, int out_size, void* d_ws, size_t ws_size,
                              hipStream_t stream) {
  const float* x      = (const float*)d_in[0];
  const float* sinb   = (const float*)d_in[1];
  const float* cosb   = (const float*)d_in[2];
  const float* mask   = (const float*)d_in[3];
  const float* pos_w  = (const float*)d_in[4];
  const float* pos_b  = (const float*)d_in[5];
  const float* ln1_g  = (const float*)d_in[6];
  const float* ln1_b  = (const float*)d_in[7];
  const float* q_w    = (const float*)d_in[8];
  const float* q_b    = (const float*)d_in[9];
  const float* k_w    = (const float*)d_in[10];
  const float* k_b    = (const float*)d_in[11];
  const float* v_w    = (const float*)d_in[12];
  const float* v_b    = (const float*)d_in[13];
  const float* lepe_w = (const float*)d_in[14];
  const float* lepe_b = (const float*)d_in[15];
  const float* out_w  = (const float*)d_in[16];
  const float* out_b  = (const float*)d_in[17];
  const float* ln2_g  = (const float*)d_in[18];
  const float* ln2_b  = (const float*)d_in[19];
  const float* fc1_w  = (const float*)d_in[20];
  const float* fc1_b  = (const float*)d_in[21];
  const float* fc2_w  = (const float*)d_in[22];
  const float* fc2_b  = (const float*)d_in[23];
  const float* dw_w   = (const float*)d_in[24];
  const float* dw_b   = (const float*)d_in[25];
  float* out = (float*)d_out;

  const int Bn = 4, H = 32, W = 32, C = 384, FFN = 1536, NC3 = 1152, M = 4096;
  const float scaling = 0.17677669529663687f;   // HD^-0.5

  char* ws = (char*)d_ws;
  size_t off = 0;
  auto alloc = [&](size_t bytes) -> void* {
    void* p = ws + off;
    off += (bytes + 255) & ~(size_t)255;
    return p;
  };
  float* x1   = (float*)alloc((size_t)M * C * 4);
  u16*   xn1  = (u16*)  alloc((size_t)M * C * 2);
  u16*   wqkv = (u16*)  alloc((size_t)C * NC3 * 2);   // transposed [1152][384]
  float* bqkv = (float*)alloc((size_t)NC3 * 4);
  float* qkv  = (float*)alloc((size_t)M * NC3 * 4);
  u16*   qrb  = (u16*)  alloc((size_t)M * C * 2);
  u16*   krb  = (u16*)  alloc((size_t)M * C * 2);
  u16*   vTb  = (u16*)  alloc((size_t)M * C * 2);
  float* lepe = (float*)alloc((size_t)M * C * 4);
  u16*   obf  = (u16*)  alloc((size_t)M * C * 2);
  u16*   wout = (u16*)  alloc((size_t)C * C * 2);     // transposed [384][384]
  float* x2   = (float*)alloc((size_t)M * C * 4);
  u16*   xn2  = (u16*)  alloc((size_t)M * C * 2);
  u16*   wfc1 = (u16*)  alloc((size_t)C * FFN * 2);   // transposed [1536][384]
  float* hbuf = (float*)alloc((size_t)M * FFN * 4);
  u16*   hbf  = (u16*)  alloc((size_t)M * FFN * 2);
  u16*   wfc2 = (u16*)  alloc((size_t)FFN * C * 2);   // transposed [384][1536]
  (void)ws_size; (void)n_in; (void)in_sizes; (void)out_size; (void)Bn;

  auto blocks = [](int n) { return (n + 255) / 256; };

  // Weight conversion + transpose (independent of activations)
  convert_wT_kernel<<<blocks(C * C), 256, 0, stream>>>(q_w,  wqkv, C, C, 0,   1.0f);
  convert_wT_kernel<<<blocks(C * C), 256, 0, stream>>>(k_w,  wqkv, C, C, 384, scaling);
  convert_wT_kernel<<<blocks(C * C), 256, 0, stream>>>(v_w,  wqkv, C, C, 768, 1.0f);
  pack_bias_kernel<<<blocks(C), 256, 0, stream>>>(q_b, bqkv, C, 0,   1.0f);
  pack_bias_kernel<<<blocks(C), 256, 0, stream>>>(k_b, bqkv, C, 384, scaling);
  pack_bias_kernel<<<blocks(C), 256, 0, stream>>>(v_b, bqkv, C, 768, 1.0f);
  convert_wT_kernel<<<blocks(C * C),   256, 0, stream>>>(out_w, wout, C, C,   0, 1.0f);
  convert_wT_kernel<<<blocks(C * FFN), 256, 0, stream>>>(fc1_w, wfc1, C, FFN, 0, 1.0f);
  convert_wT_kernel<<<blocks(FFN * C), 256, 0, stream>>>(fc2_w, wfc2, FFN, C, 0, 1.0f);

  // 1) x1 = x + dwconv3x3(x)
  dwconv_kernel<<<blocks(M * C), 256, 0, stream>>>(x, pos_w, pos_b, x, x1, nullptr,
                                                   M * C, H, W, C, 1, C, 0);
  // 2) xn1 = LN(x1)  (bf16)
  ln_kernel<<<M, 128, 0, stream>>>(x1, ln1_g, ln1_b, xn1);
  // 3) qkv = xn1 @ [q|k*s|v] + bias
  gemm_kernel<<<dim3(M / 64, NC3 / 128), 256, 0, stream>>>(xn1, wqkv, bqkv, nullptr,
                                                           qkv, M, NC3, C, 0);
  // 4) rotary + layout for attention
  rotary_kernel<<<blocks(M * 12 * 16), 256, 0, stream>>>(qkv, sinb, cosb, qrb, krb, vTb);
  // 5) lepe = dwconv5x5(v)
  dwconv_kernel<<<blocks(M * C), 256, 0, stream>>>(qkv, lepe_w, lepe_b, nullptr, lepe, nullptr,
                                                   M * C, H, W, C, 2, NC3, 768);
  // 6) fused attention -> obf = bf16(softmax(qk+mask)@v + lepe)
  attn_kernel<<<dim3(64, 12, 4), 32, 0, stream>>>(qrb, krb, vTb, mask, lepe, obf);
  // 7) x2 = x1 + obf @ out_w + out_b
  gemm_kernel<<<dim3(M / 64, C / 128), 256, 0, stream>>>(obf, wout, out_b, x1, x2, M, C, C, 0);
  // 8) xn2 = LN(x2)
  ln_kernel<<<M, 128, 0, stream>>>(x2, ln2_g, ln2_b, xn2);
  // 9) h = gelu(xn2 @ fc1 + b)
  gemm_kernel<<<dim3(M / 64, FFN / 128), 256, 0, stream>>>(xn2, wfc1, fc1_b, nullptr,
                                                           hbuf, M, FFN, C, 1);
  // 10) hbf = bf16(h + dwconv3x3(h))
  dwconv_kernel<<<blocks(M * FFN), 256, 0, stream>>>(hbuf, dw_w, dw_b, hbuf, nullptr, hbf,
                                                     M * FFN, H, W, FFN, 1, FFN, 0);
  // 11) out = x2 + hbf @ fc2 + b
  gemm_kernel<<<dim3(M / 64, C / 128), 256, 0, stream>>>(hbf, wfc2, fc2_b, x2, out, M, C, FFN, 0);
}
